// SAGE_1168231104600
// MI455X (gfx1250) — matrix operand order repaired and checked by reading, not compiled, not run on hardware
//
#include <hip/hip_runtime.h>
#include <hip/hip_bf16.h>

typedef __attribute__((ext_vector_type(16))) _Float16 v16h;
typedef __attribute__((ext_vector_type(4)))  _Float16 v4h;
typedef __attribute__((ext_vector_type(8)))  float    v8f;

// ---------------------------------------------------------------------------
// Utility kernels
// ---------------------------------------------------------------------------

__global__ __launch_bounds__(256) void zero_f32(float* __restrict__ p, long long n) {
    long long i = (long long)blockIdx.x * blockDim.x + threadIdx.x;
    if (i < n) p[i] = 0.0f;
}

__global__ __launch_bounds__(256) void deg_count(const long long* __restrict__ dst,
                                                 float* __restrict__ deg, long long E) {
    long long e = (long long)blockIdx.x * blockDim.x + threadIdx.x;
    if (e < E) atomicAdd(&deg[dst[e]], 1.0f);
}

__global__ __launch_bounds__(256) void recip_deg(float* __restrict__ deg, int n) {
    int i = blockIdx.x * blockDim.x + threadIdx.x;
    if (i < n) deg[i] = 1.0f / fmaxf(deg[i], 1.0f);
}

// Scatter-add feat[src[e], :] into msg[dst[e], :]; 4 channels per thread.
__global__ __launch_bounds__(256) void scatter_add4(const float* __restrict__ feat,
                                                    const long long* __restrict__ src,
                                                    const long long* __restrict__ dst,
                                                    float* __restrict__ msg,
                                                    int C, long long E) {
    const int vpr = C >> 2;
    long long t = (long long)blockIdx.x * blockDim.x + threadIdx.x;
    long long total = E * (long long)vpr;
    if (t >= total) return;
    long long e  = t / vpr;
    int       c4 = (int)(t - e * vpr);
    long long s = src[e];
    long long d = dst[e];
    const float4 v = ((const float4*)(feat + s * (long long)C))[c4];
    float* p = msg + d * (long long)C + (long long)c4 * 4;
    atomicAdd(p + 0, v.x);
    atomicAdd(p + 1, v.y);
    atomicAdd(p + 2, v.z);
    atomicAdd(p + 3, v.w);
}

// ---------------------------------------------------------------------------
// Pre-pack weights into f16 WMMA B-fragment order.
// Logical W_combined[KT][NC] = [Wl ; Wr].  For k-step ks, column tile t,
// lane l, element i:  K = ks*32 + i + (l>=16 ? 16 : 0), c = t*16 + (l&15).
// Packed layout: Wp[((ks*TILES + t)*32 + l)*16 + i]  -> one 32B load per lane.
// ---------------------------------------------------------------------------
template <int NC, int KA, int KT>
__global__ __launch_bounds__(256) void pack_w(const float* __restrict__ Wl,
                                              const float* __restrict__ Wr,
                                              _Float16* __restrict__ Wp) {
    constexpr int TILES = NC / 16;
    int o = blockIdx.x * blockDim.x + threadIdx.x;
    if (o >= KT * NC) return;
    int i    = o & 15;
    int l    = (o >> 4) & 31;
    int rest = o >> 9;
    int t    = rest % TILES;
    int ks   = rest / TILES;
    int kg   = ks * 32 + i + ((l >= 16) ? 16 : 0);
    int c    = t * 16 + (l & 15);
    float v  = (kg < KA) ? Wl[kg * NC + c] : Wr[(kg - KA) * NC + c];
    Wp[o] = (_Float16)v;
}

// ---------------------------------------------------------------------------
// Fused SAGE GEMM: out[n,0:NC] = [ msg[n,:]*rdeg[n] | xin[n,:] ] @ [Wl;Wr] + bias
// A staged in LDS in fragment order (f16); B read pre-packed (f16, coalesced).
// ---------------------------------------------------------------------------
template <int NC, int KA, int KX, int WAVES, int TPW, bool RELU>
__global__ __launch_bounds__(WAVES * 32) void sage_gemm(
        const float* __restrict__ msg, const float* __restrict__ rdeg,
        const float* __restrict__ xin,
        const _Float16* __restrict__ Wp,
        const float* __restrict__ bias,
        float* __restrict__ out, int nrows) {
    static_assert(KA == KX, "agg and self feature widths must match");
    constexpr int KT    = KA + KX;
    constexpr int TILES = NC / 16;

    __shared__ __align__(32) _Float16 Ah[32][16];  // [lane][elem], fragment order

    const int tid  = threadIdx.x;
    const int lane = tid & 31;
    const int wave = tid >> 5;
    const int m0   = blockIdx.x * 16;
    const int l15  = lane & 15;
    const bool hi  = lane >= 16;

    v8f acc[TPW] = {};

    for (int kb = 0; kb < KT; kb += 32) {
        __syncthreads();
        // Uniform per k-block: agg part (scaled) or self part.
        const bool  agg  = (kb < KA);
        const float* bp  = agg ? msg : xin;
        const int    ko  = agg ? kb : (kb - KA);
        // Cooperative load: 16 rows x 32 k as float4, scatter to LDS as f16
        // already in A-fragment order.
        for (int i4 = tid; i4 < 128; i4 += WAVES * 32) {
            int r  = i4 >> 3;
            int kc = (i4 & 7) * 4;           // 0,4,...,28
            int node = m0 + r;
            if (node >= nrows) node = nrows - 1;
            float4 v = *(const float4*)(bp + (long long)node * KA + ko + kc);
            if (agg) {
                float s = rdeg[node];
                v.x *= s; v.y *= s; v.z *= s; v.w *= s;
            }
            const bool hi2  = (kc & 16) != 0;
            const int  ln2  = r + (hi2 ? 16 : 0);
            const int  i0   = kc & 15;
            v4h hv;
            hv[0] = (_Float16)v.x; hv[1] = (_Float16)v.y;
            hv[2] = (_Float16)v.z; hv[3] = (_Float16)v.w;
            *(v4h*)&Ah[ln2][i0] = hv;
        }
        __syncthreads();

        // A fragment: one contiguous 32B LDS read per lane.
        const v16h a = *(const v16h*)Ah[lane];

        // B fragments: one coalesced 32B global read per lane per tile.
        const int ks = kb >> 5;
#pragma unroll
        for (int j = 0; j < TPW; ++j) {
            const int t = wave * TPW + j;
            const v16h b = *(const v16h*)(Wp +
                ((long long)(ks * TILES + t) * 32 + lane) * 16);
            acc[j] = __builtin_amdgcn_wmma_f32_16x16x32_f16(
                false, a, false, b, (short)0, acc[j], false, false);
        }
    }

    // Store C/D (VGPR j -> row j / j+8; lane -> column).
    const int row_off = hi ? 8 : 0;
#pragma unroll
    for (int j = 0; j < TPW; ++j) {
        const int c  = (wave * TPW + j) * 16 + l15;
        const float bc = bias[c];
#pragma unroll
        for (int i = 0; i < 8; ++i) {
            int row = m0 + row_off + i;
            if (row < nrows) {
                float v = acc[j][i] + bc;
                if (RELU) v = fmaxf(v, 0.0f);
                out[(long long)row * NC + c] = v;
            }
        }
    }
}

// ---------------------------------------------------------------------------
// Launch
// ---------------------------------------------------------------------------
extern "C" void kernel_launch(void* const* d_in, const int* in_sizes, int n_in,
                              void* d_out, int out_size, void* d_ws, size_t ws_size,
                              hipStream_t stream) {
    const float*     x   = (const float*)d_in[0];
    const long long* ei  = (const long long*)d_in[1];
    const float*     W1l = (const float*)d_in[2];
    const float*     b1  = (const float*)d_in[3];
    const float*     W1r = (const float*)d_in[4];
    const float*     W2l = (const float*)d_in[5];
    const float*     b2  = (const float*)d_in[6];
    const float*     W2r = (const float*)d_in[7];
    float*           out = (float*)d_out;

    const int       Cin = 128, Chid = 256;
    const int       N   = in_sizes[0] / Cin;
    const long long E   = (long long)in_sizes[1] / 2;
    const long long* src = ei;
    const long long* dst = ei + E;

    // Workspace: deg[N] | msg1[N*128] | msg2[N*256] | h[N*256] | Wp1 | Wp2 (f16)
    float* deg  = (float*)d_ws;
    float* msg1 = deg  + N;
    float* msg2 = msg1 + (long long)N * Cin;
    float* h    = msg2 + (long long)N * Chid;
    _Float16* Wp1 = (_Float16*)(h + (long long)N * Chid);
    _Float16* Wp2 = Wp1 + 256 * 256;   // KT1*NC1
    // Wp2 occupies 512*64 halfwords.

    const long long nzero = (long long)N * (1 + Cin + Chid);
    zero_f32<<<(int)((nzero + 255) / 256), 256, 0, stream>>>(deg, nzero);

    pack_w<256, 128, 256><<<(256 * 256 + 255) / 256, 256, 0, stream>>>(W1l, W1r, Wp1);
    pack_w<64, 256, 512><<<(512 * 64 + 255) / 256, 256, 0, stream>>>(W2l, W2r, Wp2);

    deg_count<<<(int)((E + 255) / 256), 256, 0, stream>>>(dst, deg, E);

    {
        long long tot = E * (Cin / 4);
        scatter_add4<<<(int)((tot + 255) / 256), 256, 0, stream>>>(x, src, dst, msg1, Cin, E);
    }

    recip_deg<<<(N + 255) / 256, 256, 0, stream>>>(deg, N);

    const int mblocks = (N + 15) / 16;
    // Layer 1: NC=256, K=128+128, 8 waves x 2 tiles, ReLU.
    sage_gemm<256, 128, 128, 8, 2, true><<<mblocks, 256, 0, stream>>>(
        msg1, deg, x, Wp1, b1, h, N);

    {
        long long tot = E * (Chid / 4);
        scatter_add4<<<(int)((tot + 255) / 256), 256, 0, stream>>>(h, src, dst, msg2, Chid, E);
    }

    // Layer 2: NC=64, K=256+256, 4 waves x 1 tile, no ReLU.
    sage_gemm<64, 256, 256, 4, 1, false><<<mblocks, 128, 0, stream>>>(
        msg2, deg, h, Wp2, b2, out, N);
}
